// RPN_53145925320996
// MI455X (gfx1250) — compile-verified, matrix operand order checked by reading, not run on hardware
//
#include <hip/hip_runtime.h>
#include <cmath>

// ---------------- problem constants ----------------
#define BATCH   4
#define DIN     1024
#define CH      512
#define Hf      50
#define Wf      76
#define HP      52      // padded H
#define WP      78      // padded W
#define HWPAD   (HP*WP)              // 4056
#define NSPAT   (BATCH*Hf*Wf)        // 15200
#define NPAD    15360                // round up to 128
#define KCONV   (9*DIN)              // 9216
#define ANCH    9
#define NANCH   (Hf*Wf*ANCH)         // 34200 per image
#define POST_NMS 300
#define NMS_TH  0.7f

typedef __attribute__((ext_vector_type(16))) __bf16 v16bf;
typedef __attribute__((ext_vector_type(8)))  float  v8f;
typedef __attribute__((ext_vector_type(4)))  unsigned int u32x4;
typedef __attribute__((ext_vector_type(8)))  int  i32x8;
typedef __attribute__((ext_vector_type(4)))  int  i32x4;

__device__ __forceinline__ unsigned short f2bf_bits(float f) {
    union { float f; unsigned u; } c; c.f = f;
    unsigned u = c.u;
    if ((u & 0x7fffffffu) > 0x7f800000u) return (unsigned short)((u >> 16) | 0x40u); // NaN
    u += 0x7fffu + ((u >> 16) & 1u);   // round-to-nearest-even
    return (unsigned short)(u >> 16);
}

// ------------- 1) pad + convert feature map to bf16 -------------
__global__ __launch_bounds__(256) void k_pad_feat(const float* __restrict__ feat,
                                                  unsigned short* __restrict__ out) {
    int idx = blockIdx.x * 256 + threadIdx.x;          // over 4*1024*52*78
    int pw = idx % WP; int t = idx / WP;
    int ph = t % HP;  t /= HP;
    int c  = t % DIN; int b = t / DIN;
    float v = 0.f;
    if (ph >= 1 && ph <= Hf && pw >= 1 && pw <= Wf)
        v = feat[(((size_t)b*DIN + c)*Hf + (ph-1))*Wf + (pw-1)];
    out[idx] = f2bf_bits(v);
}

// ------------- 2) transpose + convert conv weights: wT[co][tap][ci] -------------
__global__ __launch_bounds__(256) void k_conv_w(const float* __restrict__ w,
                                                unsigned short* __restrict__ out) {
    int idx = blockIdx.x * 256 + threadIdx.x;          // over 512*9216
    int r  = idx % KCONV; int co = idx / KCONV;
    int tap = r / DIN;    int ci = r % DIN;
    out[idx] = f2bf_bits(w[((size_t)co*DIN + ci)*9 + tap]);
}

// ------------- 3) pack 1x1 head weights into 64x512 bf16 + bias -------------
__global__ __launch_bounds__(256) void k_head_w(const float* __restrict__ cls_w,
                                                const float* __restrict__ cls_b,
                                                const float* __restrict__ bbox_w,
                                                const float* __restrict__ bbox_b,
                                                unsigned short* __restrict__ A2,
                                                float* __restrict__ b2) {
    int idx = blockIdx.x * 256 + threadIdx.x;          // over 64*512
    if (idx >= 64*512) return;
    int c = idx & 511; int o = idx >> 9;
    float v = 0.f;
    if (o < 18)      v = cls_w[o*512 + c];
    else if (o < 54) v = bbox_w[(o-18)*512 + c];
    A2[idx] = f2bf_bits(v);
    if (idx < 64) {
        float bv = 0.f;
        if (idx < 18)      bv = cls_b[idx];
        else if (idx < 54) bv = bbox_b[idx-18];
        b2[idx] = bv;
    }
}

// ------------- 4) 3x3 conv as implicit GEMM with WMMA bf16 -------------
// grid: (512/128, NPAD/128), block 256 (8 waves)
// block tile 128(M=co) x 128(N=spatial); wave tile 32x64 -> 8 WMMAs per K-step
__global__ __launch_bounds__(256) void k_conv_wmma(
    const unsigned short* __restrict__ feat,   // [4][1024][52][78] bf16, zero padded
    const unsigned short* __restrict__ wT,     // [512][9][1024]    bf16
    const float* __restrict__ bias,            // [512]
    unsigned short* __restrict__ xout)         // [512][NPAD]       bf16
{
    __shared__ __align__(16) unsigned short As[128][40];  // [co][k], padded row (20 dwords)
    __shared__ __align__(16) unsigned short Bs[128][40];  // [n][k],  padded row

    const int tid  = threadIdx.x;
    const int lane = tid & 31;
    const int wv   = tid >> 5;
    const int mwv  = wv & 3;        // M group of 32: rows mwv*32 .. +31
    const int nwv  = wv >> 2;       // N group of 64: cols nwv*64 .. +63
    const int fm   = lane & 15;
    const int kh2  = lane >> 4;

    const int co0 = blockIdx.x << 7;
    const int n0  = blockIdx.y << 7;

    // B gather coordinates: thread owns n column ln, k rows kq*16..kq*16+15
    const int ln = tid & 127;
    const int kq = tid >> 7;              // 0 or 1
    const int bn = n0 + ln;
    int fbase = 0;
    const bool vld = (bn < NSPAT);
    if (vld) {
        int bb = bn / (Hf*Wf); int r = bn - bb*(Hf*Wf);
        int bh = r / Wf;       int bw = r - bh*Wf;
        fbase = bb*DIN*HWPAD + bh*WP + bw;
    }

    // A load coordinates: thread loads 32B (2x float4) of one weight row
    const int ar = tid >> 1;              // 0..127
    const int ac = (tid & 1) << 4;        // 0 or 16
    const unsigned short* aptr = wT + (size_t)(co0 + ar)*KCONV + ac;

    v8f acc[2][4];
    #pragma unroll
    for (int i = 0; i < 2; ++i)
        #pragma unroll
        for (int j = 0; j < 4; ++j)
            acc[i][j] = (v8f){0.f,0.f,0.f,0.f,0.f,0.f,0.f,0.f};

    for (int k0 = 0; k0 < KCONV; k0 += 32) {
        const int tap = k0 >> 10;
        const int ci0 = k0 & 1023;
        const int dh = tap / 3, dw = tap - dh*3;

        // stage A (weights): coalesced 16B loads
        float4 a0 = *(const float4*)(aptr + k0);
        float4 a1 = *(const float4*)(aptr + k0 + 8);
        *(float4*)(&As[ar][ac])     = a0;
        *(float4*)(&As[ar][ac + 8]) = a1;
        if (k0 + 32 < KCONV)
            __builtin_prefetch((const void*)(aptr + k0 + 32), 0, 1);

        // stage B (features): coalesced-across-lanes u16 gathers
        const int gbase = fbase + (ci0 + kq*16)*HWPAD + dh*WP + dw;
        #pragma unroll
        for (int j = 0; j < 16; ++j) {
            unsigned short vB = 0;
            if (vld) vB = feat[gbase + j*HWPAD];
            Bs[ln][kq*16 + j] = vB;
        }
        __syncthreads();

        // build fragments (ISA 7.12.2 16-bit A layout; B symmetric; rows at 20-dword stride)
        union Frag { v16bf v; unsigned u[8]; } af[2], bf[4];
        #pragma unroll
        for (int v = 0; v < 8; ++v) {
            const int kk = ((v >> 2) << 4) + (kh2 << 3) + ((v & 3) << 1);
            #pragma unroll
            for (int i = 0; i < 2; ++i)
                af[i].u[v] = *(const unsigned*)(&As[(mwv<<5) + (i<<4) + fm][kk]);
            #pragma unroll
            for (int j = 0; j < 4; ++j)
                bf[j].u[v] = *(const unsigned*)(&Bs[(nwv<<6) + (j<<4) + fm][kk]);
        }
        #pragma unroll
        for (int i = 0; i < 2; ++i)
            #pragma unroll
            for (int j = 0; j < 4; ++j)
                acc[i][j] = __builtin_amdgcn_wmma_f32_16x16x32_bf16(
                    false, af[i].v, false, bf[j].v, (short)0, acc[i][j], false, false);
        __syncthreads();
    }

    // store: bias + ReLU -> bf16 x[c][n]
    #pragma unroll
    for (int i = 0; i < 2; ++i) {
        #pragma unroll
        for (int v = 0; v < 8; ++v) {
            const int co = co0 + (mwv<<5) + (i<<4) + v + (kh2<<3);
            const float bv = bias[co];
            #pragma unroll
            for (int j = 0; j < 4; ++j) {
                const int nn = n0 + (nwv<<6) + (j<<4) + fm;
                float val = acc[i][j][v] + bv;
                val = val > 0.f ? val : 0.f;
                xout[(size_t)co*NPAD + nn] = f2bf_bits(val);
            }
        }
    }
}

// ------------- 5) 1x1 head GEMM (cls+bbox fused, M=64) with WMMA + TDM -------------
// grid: (1, NPAD/32), block 256. A2 (64x512 bf16) pulled into LDS once via
// tensor_load_to_lds with pad_enable (4 dwords every 256 dwords -> 520-elem rows).
__global__ __launch_bounds__(256) void k_head_wmma(
    const unsigned short* __restrict__ xbf,  // [512][NPAD] bf16
    const unsigned short* __restrict__ A2,   // [64][512]   bf16
    const float* __restrict__ b2,            // [64]
    float* __restrict__ out2)                // [64][NPAD]  f32
{
    __shared__ __align__(16) unsigned short As2[64][520]; // 64 rows, 1040B stride
    __shared__ __align__(16) unsigned short Bs[32][40];

    const int tid  = threadIdx.x;
    const int lane = tid & 31;
    const int wv   = tid >> 5;
    const int mw   = wv & 3;
    const int nw   = wv >> 2;
    const int fm   = lane & 15;
    const int kh2  = lane >> 4;

    // ---- TDM: wave 0 DMAs the whole A matrix into LDS ----
    if (wv == 0) {
        unsigned lds_addr = (unsigned)(size_t)(&As2[0][0]);   // LDS aperture: low 32 bits = LDS offset
        unsigned long long ga = (unsigned long long)(size_t)A2;
        u32x4 g0;
        g0[0] = 1u;                                           // count=1, user mode
        g0[1] = lds_addr;                                     // bits 63:32
        g0[2] = (unsigned)ga;                                 // global_addr[31:0]
        g0[3] = (unsigned)(ga >> 32) | 0x80000000u;           // global_addr[56:32] | type=2
        i32x8 g1;
        g1[0] = (1 << 16)      // data_size = 2 bytes
              | (1 << 20)      // pad_enable
              | (7 << 22)      // pad_interval: every 256 dwords (= one 512-elem row)
              | (3 << 25);     // pad_amount: 4 dwords
        g1[1] = (int)(512u << 16);   // tensor_dim0 = 512
        g1[2] = (int)(64u  << 16);   // tensor_dim1 = 64
        g1[3] = (int)(512u << 16);   // tile_dim0 = 512
        g1[4] = 64;                  // tile_dim1 = 64, tile_dim2 = 0
        g1[5] = 512;                 // tensor_dim0_stride = 512
        g1[6] = 0;
        g1[7] = 0;
        i32x4 z4 = {0, 0, 0, 0};
        i32x8 z8 = {0, 0, 0, 0, 0, 0, 0, 0};
        __builtin_amdgcn_tensor_load_to_lds(g0, g1, z4, z4, z8, 0);
        __builtin_amdgcn_s_wait_tensorcnt(0);
    }
    __syncthreads();

    const int n0 = blockIdx.y << 5;
    const int ln = tid & 31;
    const int bn = n0 + ln;
    const int krow0 = tid >> 5;

    v8f acc = {0.f,0.f,0.f,0.f,0.f,0.f,0.f,0.f};

    for (int k0 = 0; k0 < 512; k0 += 32) {
        #pragma unroll
        for (int j = 0; j < 4; ++j) {
            int kr = krow0 + (j << 3);
            Bs[ln][kr] = xbf[(size_t)(k0 + kr)*NPAD + bn];
        }
        __syncthreads();

        union Frag { v16bf v; unsigned u[8]; } af, bfrag;
        #pragma unroll
        for (int v = 0; v < 8; ++v) {
            const int kk = ((v >> 2) << 4) + (kh2 << 3) + ((v & 3) << 1);
            af.u[v]    = *(const unsigned*)(&As2[(mw<<4) + fm][k0 + kk]);
            bfrag.u[v] = *(const unsigned*)(&Bs[(nw<<4) + fm][kk]);
        }
        acc = __builtin_amdgcn_wmma_f32_16x16x32_bf16(false, af.v, false, bfrag.v,
                                                      (short)0, acc, false, false);
        __syncthreads();
    }

    const int co = (mw<<4);
    const int nn = n0 + (nw<<4) + fm;
    #pragma unroll
    for (int v = 0; v < 8; ++v) {
        int m = v + (kh2 << 3);
        out2[(size_t)(co + m)*NPAD + nn] = acc[v] + b2[co + m];
    }
}

// ------------- 6) softmax + anchor decode + clip + min-size filter -------------
struct AnchorBases { float v[36]; };

__global__ __launch_bounds__(256) void k_decode(
    const float* __restrict__ out2,     // [64][NPAD]
    const float* __restrict__ im_info,  // [4][3]
    AnchorBases ab,
    float* __restrict__ scores, float* __restrict__ smut,
    float* __restrict__ boxes)
{
    int idx = blockIdx.x * 256 + threadIdx.x;
    if (idx >= BATCH*NANCH) return;
    int b = idx / NANCH;
    int i = idx - b*NANCH;
    int a = i % ANCH;
    int cell = i / ANCH;
    int w = cell % Wf;
    int h = cell / Wf;
    int n = b*(Hf*Wf) + cell;

    float c0 = out2[(size_t)a*NPAD + n];
    float c1 = out2[(size_t)(ANCH + a)*NPAD + n];
    float sc = 1.f / (1.f + __expf(c0 - c1));   // softmax fg prob

    float d0 = out2[(size_t)(18 + a*4 + 0)*NPAD + n];
    float d1 = out2[(size_t)(18 + a*4 + 1)*NPAD + n];
    float d2 = out2[(size_t)(18 + a*4 + 2)*NPAD + n];
    float d3 = out2[(size_t)(18 + a*4 + 3)*NPAD + n];

    float sx = w * 16.f, sy = h * 16.f;
    float ax1 = ab.v[a*4+0] + sx, ay1 = ab.v[a*4+1] + sy;
    float ax2 = ab.v[a*4+2] + sx, ay2 = ab.v[a*4+3] + sy;
    float aw = ax2 - ax1 + 1.f, ah = ay2 - ay1 + 1.f;
    float axc = ax1 + 0.5f*aw,  ayc = ay1 + 0.5f*ah;

    float cx = axc + d0*aw, cy = ayc + d1*ah;
    float pw = __expf(d2)*aw, ph = __expf(d3)*ah;
    float x1 = cx - 0.5f*pw, y1 = cy - 0.5f*ph;
    float x2 = cx + 0.5f*pw, y2 = cy + 0.5f*ph;

    float ih = im_info[b*3+0], iw = im_info[b*3+1], iscale = im_info[b*3+2];
    x1 = fminf(fmaxf(x1, 0.f), iw - 1.f);  x2 = fminf(fmaxf(x2, 0.f), iw - 1.f);
    y1 = fminf(fmaxf(y1, 0.f), ih - 1.f);  y2 = fminf(fmaxf(y2, 0.f), ih - 1.f);

    float ws = x2 - x1 + 1.f, hs = y2 - y1 + 1.f;
    float ms = 16.f * iscale;
    if (!(ws >= ms && hs >= ms)) sc = -__builtin_inff();

    scores[idx] = sc;
    smut[idx]   = sc;
    boxes[(size_t)idx*4+0] = x1; boxes[(size_t)idx*4+1] = y1;
    boxes[(size_t)idx*4+2] = x2; boxes[(size_t)idx*4+3] = y2;
}

// ------------- 7) greedy NMS: one block per image, 300 sequential rounds -------------
__global__ __launch_bounds__(1024) void k_nms(
    const float* __restrict__ scores, float* __restrict__ smut,
    const float* __restrict__ boxes,
    float* __restrict__ out_scores,   // [4][300]
    float* __restrict__ out_boxes)    // [4][300][4]
{
    __shared__ float svals[1024];
    __shared__ int   sidx[1024];
    __shared__ float jbox[4];

    const int b = blockIdx.x, tid = threadIdx.x;
    const float* sc = scores + (size_t)b*NANCH;
    float*       m  = smut   + (size_t)b*NANCH;
    const float* bx = boxes  + (size_t)b*NANCH*4;
    const float NEG = -__builtin_inff();

    for (int it = 0; it < POST_NMS; ++it) {
        // ---- parallel argmax (first-max tie-break, like jnp.argmax) ----
        float best = NEG; int bi = 0x7fffffff;
        for (int i = tid; i < NANCH; i += 1024) {
            float v = m[i];
            if (v > best || (v == best && i < bi)) { best = v; bi = i; }
        }
        svals[tid] = best; sidx[tid] = bi;
        __syncthreads();
        for (int s = 512; s > 0; s >>= 1) {
            if (tid < s) {
                float ov = svals[tid+s]; int oi = sidx[tid+s];
                if (ov > svals[tid] || (ov == svals[tid] && oi < sidx[tid])) {
                    svals[tid] = ov; sidx[tid] = oi;
                }
            }
            __syncthreads();
        }
        if (tid == 0) {
            int j = sidx[0];
            jbox[0] = bx[(size_t)j*4+0]; jbox[1] = bx[(size_t)j*4+1];
            jbox[2] = bx[(size_t)j*4+2]; jbox[3] = bx[(size_t)j*4+3];
            out_scores[b*POST_NMS + it] = sc[j];
            out_boxes[((size_t)b*POST_NMS + it)*4+0] = jbox[0];
            out_boxes[((size_t)b*POST_NMS + it)*4+1] = jbox[1];
            out_boxes[((size_t)b*POST_NMS + it)*4+2] = jbox[2];
            out_boxes[((size_t)b*POST_NMS + it)*4+3] = jbox[3];
        }
        __syncthreads();

        const float jx1 = jbox[0], jy1 = jbox[1], jx2 = jbox[2], jy2 = jbox[3];
        const float jarea = (jx2 - jx1 + 1.f) * (jy2 - jy1 + 1.f);
        for (int i = tid; i < NANCH; i += 1024) {
            float ix1 = bx[(size_t)i*4+0], iy1 = bx[(size_t)i*4+1];
            float ix2 = bx[(size_t)i*4+2], iy2 = bx[(size_t)i*4+3];
            float xx1 = fmaxf(jx1, ix1), yy1 = fmaxf(jy1, iy1);
            float xx2 = fminf(jx2, ix2), yy2 = fminf(jy2, iy2);
            float inter = fmaxf(0.f, xx2 - xx1 + 1.f) * fmaxf(0.f, yy2 - yy1 + 1.f);
            float iarea = (ix2 - ix1 + 1.f) * (iy2 - iy1 + 1.f);
            float iou = inter / (jarea + iarea - inter);
            if (iou > NMS_TH) m[i] = NEG;
        }
        __syncthreads();
    }
}

// ---------------- host: anchor bases (round-half-even matches np.round) ----------------
static void gen_anchor_bases(AnchorBases& ab) {
    const double ratios[3] = {0.5, 1.0, 2.0};
    const double scales[3] = {8.0, 16.0, 32.0};
    double w = 16.0, h = 16.0;
    double xc = 0.5*(w - 1.0), yc = 0.5*(h - 1.0);
    double size = w*h;
    int idx = 0;
    for (int ri = 0; ri < 3; ++ri) {
        double ws = nearbyint(sqrt(size / ratios[ri]));
        double hs = nearbyint(ws * ratios[ri]);
        double rx1 = xc - 0.5*(ws - 1.0), ry1 = yc - 0.5*(hs - 1.0);
        double rx2 = xc + 0.5*(ws - 1.0), ry2 = yc + 0.5*(hs - 1.0);
        double rw = rx2 - rx1 + 1.0, rh = ry2 - ry1 + 1.0;
        double rxc = rx1 + 0.5*(rw - 1.0), ryc = ry1 + 0.5*(rh - 1.0);
        for (int si = 0; si < 3; ++si) {
            double sw = rw * scales[si], sh = rh * scales[si];
            ab.v[idx*4+0] = (float)(rxc - 0.5*(sw - 1.0));
            ab.v[idx*4+1] = (float)(ryc - 0.5*(sh - 1.0));
            ab.v[idx*4+2] = (float)(rxc + 0.5*(sw - 1.0));
            ab.v[idx*4+3] = (float)(ryc + 0.5*(sh - 1.0));
            ++idx;
        }
    }
}

extern "C" void kernel_launch(void* const* d_in, const int* in_sizes, int n_in,
                              void* d_out, int out_size, void* d_ws, size_t ws_size,
                              hipStream_t stream) {
    const float* base_feat = (const float*)d_in[0];
    const float* im_info   = (const float*)d_in[1];
    // d_in[2] gt_boxes, d_in[3] num_boxes unused
    const float* conv_w = (const float*)d_in[4];
    const float* conv_b = (const float*)d_in[5];
    const float* cls_w  = (const float*)d_in[6];
    const float* cls_b  = (const float*)d_in[7];
    const float* bbox_w = (const float*)d_in[8];
    const float* bbox_b = (const float*)d_in[9];
    float* out = (float*)d_out;

    // workspace carve-up (256B aligned)
    size_t off = 0;
    auto carve = [&](size_t bytes) {
        void* p = (char*)d_ws + off;
        off += (bytes + 255) & ~(size_t)255;
        return p;
    };
    unsigned short* feat_bf = (unsigned short*)carve((size_t)BATCH*DIN*HP*WP*2);
    unsigned short* wT      = (unsigned short*)carve((size_t)CH*KCONV*2);
    unsigned short* xbf     = (unsigned short*)carve((size_t)CH*NPAD*2);
    unsigned short* A2      = (unsigned short*)carve((size_t)64*512*2);
    float*          b2      = (float*)carve(64*4);
    float*          out2    = (float*)carve((size_t)64*NPAD*4);
    float*          scores  = (float*)carve((size_t)BATCH*NANCH*4);
    float*          smut    = (float*)carve((size_t)BATCH*NANCH*4);
    float*          boxes   = (float*)carve((size_t)BATCH*NANCH*4*4);

    AnchorBases ab;
    gen_anchor_bases(ab);

    // 1) pad+convert features
    k_pad_feat<<<(BATCH*DIN*HP*WP)/256, 256, 0, stream>>>(base_feat, feat_bf);
    // 2) transpose+convert conv weights
    k_conv_w<<<(CH*KCONV)/256, 256, 0, stream>>>(conv_w, wT);
    // 3) head weights
    k_head_w<<<(64*512)/256, 256, 0, stream>>>(cls_w, cls_b, bbox_w, bbox_b, A2, b2);
    // 4) 3x3 conv via WMMA implicit GEMM (128x128 block tile, 8 WMMA/wave/step)
    k_conv_wmma<<<dim3(CH/128, NPAD/128), 256, 0, stream>>>(feat_bf, wT, conv_b, xbf);
    // 5) fused 1x1 heads via WMMA, A staged by TDM tensor_load_to_lds
    k_head_wmma<<<dim3(1, NPAD/32), 256, 0, stream>>>(xbf, A2, b2, out2);
    // 6) decode anchors
    k_decode<<<(BATCH*NANCH + 255)/256, 256, 0, stream>>>(out2, im_info, ab, scores, smut, boxes);
    // 7) greedy NMS, outputs: scores [4][300] then boxes [4][300][4]
    k_nms<<<BATCH, 1024, 0, stream>>>(scores, smut, boxes, out, out + BATCH*POST_NMS);
}
